// GnReluConv_25400436588653
// MI455X (gfx1250) — matrix operand order repaired and checked by reading, not compiled, not run on hardware
//
#include <hip/hip_runtime.h>
#include <hip/hip_bf16.h>

// ---------------- types for CDNA5 WMMA / TDM ----------------
typedef __attribute__((ext_vector_type(16))) __bf16 v16bf;
typedef __attribute__((ext_vector_type(8)))  float  v8f;
typedef __attribute__((ext_vector_type(4)))  unsigned int u32x4;
typedef __attribute__((ext_vector_type(4)))  int    i32x4;
typedef __attribute__((ext_vector_type(8)))  int    i32x8;

union Frag16 {            // 16 bf16 = 32 bytes = two int4 chunks
    v16bf v;
    int4  q[2];
};

#define NV   500000
#define CC   64
#define FEXT 9
#define NFILT 64
#define NGRP 32
#define TILES (NV / 16)          // 31250, N is a multiple of 16
#define WVBLK 8                  // waves per block in conv kernel
#define WBYTES (FEXT * CC * NFILT * 2)   // 73728 B of bf16 weight
#define WDWORDS (WBYTES / 4)             // 18432 dwords... (2D: 1152 x 16)

// ---------------- 0: zero group accumulators ----------------
__global__ void gn_zero_kernel(float* gsum, float* gsqs) {
    int t = threadIdx.x;
    if (t < NGRP) { gsum[t] = 0.f; gsqs[t] = 0.f; }
}

// ---------------- 1: per-group sum / sumsq ----------------
__global__ void gn_stats_kernel(const float* __restrict__ lv,
                                float* __restrict__ gsum,
                                float* __restrict__ gsqs) {
    __shared__ float ssum[256];
    __shared__ float ssqs[256];
    const int c      = threadIdx.x & 63;          // channel
    const int rloc   = threadIdx.x >> 6;          // 0..3
    const int rowsPB = blockDim.x >> 6;           // 4
    int r = blockIdx.x * rowsPB + rloc;
    const int rstr = gridDim.x * rowsPB;
    float s = 0.f, s2 = 0.f;
    for (; r < NV; r += rstr) {
        float x = lv[(size_t)r * CC + c];
        s += x; s2 += x * x;
    }
    ssum[threadIdx.x] = s; ssqs[threadIdx.x] = s2;
    __syncthreads();
    if (threadIdx.x < 64) {
        s  = ssum[threadIdx.x] + ssum[threadIdx.x + 64] +
             ssum[threadIdx.x + 128] + ssum[threadIdx.x + 192];
        s2 = ssqs[threadIdx.x] + ssqs[threadIdx.x + 64] +
             ssqs[threadIdx.x + 128] + ssqs[threadIdx.x + 192];
        ssum[threadIdx.x] = s; ssqs[threadIdx.x] = s2;
    }
    __syncthreads();
    if (threadIdx.x < 64 && (threadIdx.x & 1) == 0) {
        int g = threadIdx.x >> 1;                 // C/G == 2 channels per group
        atomicAdd(&gsum[g], ssum[threadIdx.x] + ssum[threadIdx.x + 1]);
        atomicAdd(&gsqs[g], ssqs[threadIdx.x] + ssqs[threadIdx.x + 1]);
    }
}

// ---------------- 2: fold stats + gamma/beta into per-channel scale/shift ----------------
__global__ void gn_finalize_kernel(const float* __restrict__ gsum,
                                   const float* __restrict__ gsqs,
                                   const float* __restrict__ gamma,
                                   const float* __restrict__ beta,
                                   float* __restrict__ scale,
                                   float* __restrict__ shift) {
    int c = threadIdx.x;                          // 64 threads
    if (c >= CC) return;
    int g = c >> 1;
    const float invn = 1.0f / (float)((size_t)NV * (CC / NGRP));
    float mean = gsum[g] * invn;
    float var  = gsqs[g] * invn - mean * mean;
    float inv  = rsqrtf(var + 1e-5f);
    float sc   = inv * gamma[c];
    scale[c] = sc;
    shift[c] = beta[c] - mean * sc;
}

// ---------------- 3: weight transpose + fp32 -> bf16:  Wt[f][nf][c] = W[(f*64+c)*64+nf] ----------------
__global__ void wconv_kernel(const float* __restrict__ w,
                             __hip_bfloat16* __restrict__ wt) {
    int t = blockIdx.x * blockDim.x + threadIdx.x;
    if (t >= FEXT * CC * NFILT) return;
    int c  = t & 63;
    int nf = (t >> 6) & 63;
    int f  = t >> 12;
    wt[t] = __float2bfloat16(w[(f * CC + c) * NFILT + nf]);
}

// ---------------- 4: normalize + ReLU -> bf16 staging ----------------
__global__ void norm_relu_kernel(const float* __restrict__ lv,
                                 const float* __restrict__ scale,
                                 const float* __restrict__ shift,
                                 __hip_bfloat16* __restrict__ xn) {
    const size_t total = (size_t)NV * CC;
    size_t i = (size_t)blockIdx.x * blockDim.x + threadIdx.x;
    const size_t stride = (size_t)gridDim.x * blockDim.x;
    for (; i < total; i += stride) {
        int c = (int)(i & 63);
        float y = fmaxf(lv[i] * scale[c] + shift[c], 0.0f);
        xn[i] = __float2bfloat16(y);
    }
}

// ---------------- TDM helper: one DMA of the full 72 KB weight into LDS ----------------
// D# built per CDNA5 ISA ch.8: group0 = {count, lds_addr, global_addr, type=2},
// group1 = {data_size=4B, tensor 1152x16 dwords, tile 1152x16, stride 1152}.
__device__ __forceinline__ void tdm_load_weight(const __hip_bfloat16* wt, void* ldsDst) {
    unsigned long long ga = (unsigned long long)(uintptr_t)wt;
    unsigned ldsAddr = (unsigned)(uintptr_t)ldsDst;   // LDS-space byte address

    // ---- D# group 0 (128b) ----
    u32x4 g0;
    g0[0] = 1u;                                        // count=1, is_restore=0, gather off
    g0[1] = ldsAddr;                                   // lds_addr[31:0]
    g0[2] = (unsigned)(ga & 0xFFFFFFFFull);            // global_addr[31:0]
    g0[3] = ((unsigned)(ga >> 32) & 0x1FFFFFFu)        // global_addr[56:32]
            | (2u << 30);                              // type=2 ("image")

    // ---- D# group 1 (256b) ----
    const unsigned dim0 = WDWORDS / 16;                // 1152 dwords per row
    i32x8 g1;
    g1[0] = (int)(2u << 16);                           // wg_mask=0, data_size=2 (4B)
    g1[1] = (int)(dim0 << 16);                         // tensor_dim0[15:0] in [31:16]
    g1[2] = (int)(16u << 16);                          // dim0 hi=0 ; tensor_dim1=16
    g1[3] = (int)(dim0 << 16);                         // dim1 hi=0 ; tile_dim0=1152
    g1[4] = 16;                                        // tile_dim1=16, tile_dim2=0
    g1[5] = (int)dim0;                                 // tensor_dim0_stride lo
    g1[6] = 0;                                         // stride hi ; dim1_stride lo16
    g1[7] = 0;

    i32x4 z4 = {0, 0, 0, 0};
#if defined(__clang_major__) && (__clang_major__ >= 23)
    i32x8 z8 = {0, 0, 0, 0, 0, 0, 0, 0};
    __builtin_amdgcn_tensor_load_to_lds(g0, g1, z4, z4, z8, 0);
#else
    __builtin_amdgcn_tensor_load_to_lds(g0, g1, z4, z4, 0);
#endif
}

// ---------------- 5: gather + WMMA conv: out[16,64] per wave ----------------
__global__ void conv_wmma_kernel(const __hip_bfloat16* __restrict__ xn,
                                 const int* __restrict__ nidx,
                                 const __hip_bfloat16* __restrict__ wt,
                                 const float* __restrict__ bias,
                                 float* __restrict__ out) {
    extern __shared__ __hip_bfloat16 wlds[];      // 72 KB (WGP has 320 KB LDS)

    // Tensor Data Mover: wave 0 issues one DMA for the whole weight tile,
    // waits on TENSORcnt, then the block synchronizes.
    if (threadIdx.x < 32) {
        tdm_load_weight(wt, (void*)wlds);
        __builtin_amdgcn_s_wait_tensorcnt(0);
    }
    __syncthreads();

    const int wave = threadIdx.x >> 5;
    const int lane = threadIdx.x & 31;
    const int tile = blockIdx.x * WVBLK + wave;   // wave-uniform -> EXEC stays all-ones
    if (tile >= TILES) return;

    const int r0   = tile * 16;
    const int m    = lane & 15;                   // A row / D column within 16
    const int half = lane >> 4;                   // lane half selects K sub-block

    // per-lane neighbor indices for row (r0+m), all 9 taps
    int idx[FEXT];
#pragma unroll
    for (int f = 0; f < FEXT; ++f) idx[f] = nidx[(r0 + m) * FEXT + f];
#pragma unroll
    for (int f = 0; f < FEXT; ++f)
        __builtin_prefetch(xn + (size_t)idx[f] * CC, 0, 0);   // global_prefetch_b8

    // accumulators: 4 x (16x16 f32), columns j*16 + (lane%16); init with bias
    v8f acc[4];
#pragma unroll
    for (int j = 0; j < 4; ++j) {
        float bv = bias[j * 16 + m];
        acc[j] = (v8f){bv, bv, bv, bv, bv, bv, bv, bv};
    }

    const int4* wq = (const int4*)wlds;

#pragma unroll
    for (int f = 0; f < FEXT; ++f) {
        const int4* rowq = (const int4*)(xn + (size_t)idx[f] * CC);
#pragma unroll
        for (int kk = 0; kk < 2; ++kk) {          // two K=32 chunks of C=64
            // A fragment (16x32 bf16): lane half owns K = half*8..+7 and 16+half*8..+7
            Frag16 a;
            a.q[0] = rowq[kk * 4 + half];
            a.q[1] = rowq[kk * 4 + 2 + half];
#pragma unroll
            for (int j = 0; j < 4; ++j) {
                // B fragment (32x16 bf16) from LDS: Wt[f][j*16+n][kk*32 + half*8 ..+7] and [kk*32 + 16 + half*8 ..+7]
                Frag16 b;
                int base = ((f * NFILT + j * 16 + m) * 8) + kk * 4 + half;
                b.q[0] = wq[base];
                b.q[1] = wq[base + 2];
                acc[j] = __builtin_amdgcn_wmma_f32_16x16x32_bf16(
                    false, a.v, false, b.v, (short)0, acc[j], false, false);
            }
        }
    }

    // D layout: VGPR v -> row m = v + 8*half; column n = j*16 + (lane%16)
#pragma unroll
    for (int j = 0; j < 4; ++j) {
#pragma unroll
        for (int v = 0; v < 8; ++v) {
            out[(size_t)(r0 + v + 8 * half) * NFILT + j * 16 + m] = acc[j][v];
        }
    }
}

// ---------------- launcher ----------------
extern "C" void kernel_launch(void* const* d_in, const int* in_sizes, int n_in,
                              void* d_out, int out_size, void* d_ws, size_t ws_size,
                              hipStream_t stream) {
    const float* lv    = (const float*)d_in[0];
    const int*   nidx  = (const int*)d_in[1];
    const float* gamma = (const float*)d_in[2];
    const float* beta  = (const float*)d_in[3];
    const float* wgt   = (const float*)d_in[4];
    const float* bias  = (const float*)d_in[5];
    float* out = (float*)d_out;

    // workspace layout
    char* ws = (char*)d_ws;
    float* gsum  = (float*)(ws + 0);              // 32 f32
    float* gsqs  = (float*)(ws + 128);            // 32 f32
    float* scale = (float*)(ws + 256);            // 64 f32
    float* shift = (float*)(ws + 512);            // 64 f32
    __hip_bfloat16* wt = (__hip_bfloat16*)(ws + 1024);      // 36864 bf16 = 72 KB
    __hip_bfloat16* xn = (__hip_bfloat16*)(ws + 131072);    // N*C bf16 = 64 MB

    gn_zero_kernel<<<1, 64, 0, stream>>>(gsum, gsqs);
    gn_stats_kernel<<<1024, 256, 0, stream>>>(lv, gsum, gsqs);
    gn_finalize_kernel<<<1, 64, 0, stream>>>(gsum, gsqs, gamma, beta, scale, shift);
    wconv_kernel<<<(FEXT * CC * NFILT + 255) / 256, 256, 0, stream>>>(wgt, wt);
    norm_relu_kernel<<<4096, 256, 0, stream>>>(lv, scale, shift, xn);

    const int nblocks = (TILES + WVBLK - 1) / WVBLK;
    const size_t ldsbytes = (size_t)WBYTES;       // 73728
    conv_wmma_kernel<<<nblocks, 32 * WVBLK, ldsbytes, stream>>>(xn, nidx, wt, bias, out);
}
